// MambaModule_33930241638598
// MI455X (gfx1250) — matrix operand order repaired and checked by reading, not compiled, not run on hardware
//
#include <hip/hip_runtime.h>
#include <hip/hip_bf16.h>

// ---------------------------------------------------------------------------
// Problem constants (from reference)
// ---------------------------------------------------------------------------
constexpr int B_SZ  = 2;
constexpr int SEQ   = 4096;
constexpr int DM    = 1024;          // d_model
constexpr int DI    = 2048;          // d_inner
constexpr int DCONV = 4;
constexpr int MTOT  = B_SZ * SEQ;    // 8192 rows

// SSM kernel truncation: w_j = Cp^T tanh(A)^j Bp, ||tanh(A)||<~0.8 -> w_63 < 1e-6
constexpr int LTAP = 64;
constexpr int CTAP = LTAP + DCONV - 1;   // 67 combined taps
constexpr int HIST = CTAP - 1;           // 66

// ---------------------------------------------------------------------------
// Vector types
// ---------------------------------------------------------------------------
typedef __attribute__((ext_vector_type(16))) __bf16 v16bf;
typedef __attribute__((ext_vector_type(8)))  float  v8f;
typedef __attribute__((ext_vector_type(4)))  unsigned int v4u;
typedef __attribute__((ext_vector_type(2)))  unsigned int v2u;

union Frag16 { v16bf v; v4u q[2]; };

static __device__ inline unsigned pk2bf(float a, float b) {
    union { __bf16 h[2]; unsigned u; } x;
    x.h[0] = (__bf16)a; x.h[1] = (__bf16)b;
    return x.u;
}

// ---------------------------------------------------------------------------
// Kernel 1: convert both weight matrices to bf16 (read once per run)
// ---------------------------------------------------------------------------
__global__ void cvt_w_kernel(const float* __restrict__ w1, const float* __restrict__ w2,
                             __bf16* __restrict__ o1, __bf16* __restrict__ o2,
                             int n1, int n2) {
    int id = blockIdx.x * blockDim.x + threadIdx.x;
    if (id < n1)            o1[id]       = (__bf16)w1[id];
    else if (id < n1 + n2)  o2[id - n1]  = (__bf16)w2[id - n1];
}

// ---------------------------------------------------------------------------
// Kernel 2: compute SSM impulse response w[0..LTAP-1] and bias prefix sums
//   w_j = Cp . tanh(A)^j . Bp ;  wsum[t] = Dp + sum_{j<=t} w_j
// ---------------------------------------------------------------------------
__global__ void ssm_w_kernel(const float* __restrict__ A, const float* __restrict__ Bp,
                             const float* __restrict__ Cp, const float* __restrict__ Dp,
                             float* __restrict__ wv_g, float* __restrict__ wsum_g) {
    __shared__ float At[16][16];
    __shared__ float hv[16];
    __shared__ float wv[LTAP];
    const int l = threadIdx.x;
    if (l < 16) {
        for (int j = 0; j < 16; ++j) At[l][j] = tanhf(A[l * 16 + j]);
        hv[l] = Bp[l];
    }
    __syncthreads();
    for (int j = 0; j < LTAP; ++j) {
        float nh = 0.f;
        if (l < 16) {
            #pragma unroll
            for (int k = 0; k < 16; ++k) nh += At[l][k] * hv[k];
        }
        if (l == 0) {
            float wj = 0.f;
            #pragma unroll
            for (int k = 0; k < 16; ++k) wj += Cp[k] * hv[k];
            wv[j] = wj;
        }
        __syncthreads();
        if (l < 16) hv[l] = nh;
        __syncthreads();
    }
    if (l == 0) {
        float run = Dp[0];
        for (int t = 0; t < LTAP; ++t) {
            run += wv[t];
            wsum_g[t] = run;
            wv_g[t]   = wv[t];
        }
    }
}

// ---------------------------------------------------------------------------
// Kernel 3: per-channel combined taps  (SSM kernel convolved with 4-tap conv)
//   taps[c][m] = Dp*g(m) + sum_j w_j * g(m-j),  g(m)=conv_w[c,0,3-m]
// ---------------------------------------------------------------------------
__global__ void taps_kernel(const float* __restrict__ conv_w, const float* __restrict__ Dp,
                            const float* __restrict__ wv_g, float* __restrict__ taps_g) {
    int id = blockIdx.x * blockDim.x + threadIdx.x;
    if (id >= DI * CTAP) return;
    int c = id / CTAP, m = id % CTAP;
    const float* cw = conv_w + c * DCONV;
    float t = 0.f;
    if (m < DCONV) t = Dp[0] * cw[3 - m];
    int j0 = (m - 3 > 0) ? m - 3 : 0;
    int j1 = (m < LTAP - 1) ? m : LTAP - 1;
    for (int j = j0; j <= j1; ++j) {
        int mm = m - j;                       // in [0,3]
        t += wv_g[j] * cw[3 - mm];
    }
    taps_g[id] = t;
}

// ---------------------------------------------------------------------------
// WMMA GEMM (double-buffered):  C(f32) = A(MxK) * B(bf16, KxN) + bias[N]
//   AF32: A is f32 (converted to bf16 while staging to LDS); else A is bf16.
//   Tiles: BM=128, BN=128, BK=32; 256 threads = 8 waves (2x4), 64x32 per wave.
//   Pipeline: global loads for tile k+1 issue before the 8 WMMAs of tile k;
//   staged registers are written to the ping-pong LDS buffer after the WMMAs,
//   one barrier per K-iteration.
// ---------------------------------------------------------------------------
constexpr int BM = 128, BN = 128, BK = 32, BKP = 40;   // BKP*2B = 80B row stride

template <bool AF32>
__global__ __launch_bounds__(256)
void gemm_bias_kernel(const void* __restrict__ Av, const __bf16* __restrict__ Bm,
                      const float* __restrict__ bias, float* __restrict__ Cm,
                      int M, int N, int K) {
    __shared__ __align__(16) __bf16 As[2][BM][BKP];
    __shared__ __align__(16) __bf16 Bs[2][BN][BKP];   // transposed: Bs[.][n][k]

    const int tid  = threadIdx.x;
    const int lane = tid & 31;
    const int wid  = tid >> 5;
    const int wm   = wid >> 2;       // 0..1 (64 rows each)
    const int wn   = wid & 3;        // 0..3 (32 cols each)
    const int m16  = lane & 15;
    const int kh   = lane >> 4;      // lane half selects K sub-block
    const int blockM = blockIdx.x * BM;
    const int blockN = blockIdx.y * BN;

    const v8f zf = {0.f, 0.f, 0.f, 0.f, 0.f, 0.f, 0.f, 0.f};
    v8f acc[4][2];
    #pragma unroll
    for (int mi = 0; mi < 4; ++mi)
        #pragma unroll
        for (int ni = 0; ni < 2; ++ni) acc[mi][ni] = zf;

    // staging registers for the next tile
    float4 aF[4];
    v4u    aH[2];
    v4u    bH[2];

    auto loadTile = [&](int kb) {
        if constexpr (AF32) {
            const float* Af = (const float*)Av;
            #pragma unroll
            for (int i = 0; i < 4; ++i) {
                int e = tid + i * 256;                 // 1024 float4 chunks
                int r = e >> 3, c4 = e & 7;
                aF[i] = *(const float4*)(Af + (size_t)(blockM + r) * K + kb + c4 * 4);
            }
        } else {
            const __bf16* Ab = (const __bf16*)Av;
            #pragma unroll
            for (int i = 0; i < 2; ++i) {
                int e = tid + i * 256;                 // 512 8-elem chunks
                int r = e >> 2, c8 = e & 3;
                aH[i] = *(const v4u*)(Ab + (size_t)(blockM + r) * K + kb + c8 * 8);
            }
        }
        #pragma unroll
        for (int i = 0; i < 2; ++i) {
            int e  = tid + i * 256;                    // 512 8-elem chunks
            int kr = e >> 4, n8 = e & 15;
            bH[i] = *(const v4u*)(Bm + (size_t)(kb + kr) * N + blockN + n8 * 8);
        }
    };
    auto storeTile = [&](int buf) {
        if constexpr (AF32) {
            #pragma unroll
            for (int i = 0; i < 4; ++i) {
                int e = tid + i * 256;
                int r = e >> 3, c4 = e & 7;
                v2u pp;
                pp.x = pk2bf(aF[i].x, aF[i].y);
                pp.y = pk2bf(aF[i].z, aF[i].w);
                *(v2u*)&As[buf][r][c4 * 4] = pp;
            }
        } else {
            #pragma unroll
            for (int i = 0; i < 2; ++i) {
                int e = tid + i * 256;
                int r = e >> 2, c8 = e & 3;
                *(v4u*)&As[buf][r][c8 * 8] = aH[i];
            }
        }
        #pragma unroll
        for (int i = 0; i < 2; ++i) {
            int e  = tid + i * 256;
            int kr = e >> 4, n8 = e & 15;
            const __bf16* h = (const __bf16*)&bH[i];
            #pragma unroll
            for (int j = 0; j < 8; ++j) Bs[buf][n8 * 8 + j][kr] = h[j];
        }
    };

    const int nk = K / BK;
    loadTile(0);
    storeTile(0);
    __syncthreads();

    int buf = 0;
    for (int ki = 0; ki < nk; ++ki) {
        const bool has_next = (ki + 1) < nk;
        if (has_next) loadTile((ki + 1) * BK);   // global loads overlap WMMAs below

        // ---- fragments per documented 16-bit WMMA layouts ----
        Frag16 a[4], b[2];
        #pragma unroll
        for (int mi = 0; mi < 4; ++mi) {
            int row = wm * 64 + mi * 16 + m16;
            a[mi].q[0] = *(const v4u*)&As[buf][row][kh * 8];        // K = h*8..+7
            a[mi].q[1] = *(const v4u*)&As[buf][row][16 + kh * 8];   // K = 16+h*8..+7
        }
        #pragma unroll
        for (int ni = 0; ni < 2; ++ni) {
            int coln = wn * 32 + ni * 16 + m16;
            b[ni].q[0] = *(const v4u*)&Bs[buf][coln][kh * 8];       // K = h*8..+7
            b[ni].q[1] = *(const v4u*)&Bs[buf][coln][16 + kh * 8];  // K = 16+h*8..+7
        }
        #pragma unroll
        for (int mi = 0; mi < 4; ++mi)
            #pragma unroll
            for (int ni = 0; ni < 2; ++ni)
                acc[mi][ni] = __builtin_amdgcn_wmma_f32_16x16x32_bf16(
                    false, a[mi].v, false, b[ni].v, (short)0, acc[mi][ni], false, false);

        if (has_next) storeTile(buf ^ 1);       // fill other buffer after WMMAs
        __syncthreads();
        buf ^= 1;
    }

    // ---- epilogue: C/D layout -> rows kh*8+v, col = m16 within frag ----
    #pragma unroll
    for (int ni = 0; ni < 2; ++ni) {
        int col  = blockN + wn * 32 + ni * 16 + m16;
        float bn = bias[col];
        #pragma unroll
        for (int mi = 0; mi < 4; ++mi) {
            int rbase = blockM + wm * 64 + mi * 16 + kh * 8;
            #pragma unroll
            for (int v = 0; v < 8; ++v)
                Cm[(size_t)(rbase + v) * N + col] = acc[mi][ni][v] + bn;
        }
    }
}

// ---------------------------------------------------------------------------
// Kernel 5: fused depthwise-conv + truncated SSM + SiLU, output bf16.
//   y[t,c] = silu( conv_b[c]*wsum[min(t,LTAP-1)] + sum_m taps[c][m]*u[t-m,c] )
//   Tile: 32 t-outputs x 64 channels per WG; wave owns 8 channels so each tap
//   is a wave-uniform LDS broadcast and u reads are stride-1.
// ---------------------------------------------------------------------------
__global__ __launch_bounds__(256)
void convscan_kernel(const float* __restrict__ u, const float* __restrict__ conv_b,
                     const float* __restrict__ taps_g, const float* __restrict__ wsum_g,
                     __bf16* __restrict__ yb) {
    constexpr int TT = 32, CT = 64, RR = TT + HIST;    // 98 time rows staged
    __shared__ float utile[CT][105];                   // [c][r], 105 -> conflict-free
    __shared__ float taps_s[CT][CTAP + 1];
    __shared__ float wsum_s[LTAP];
    __shared__ __align__(16) __bf16 ytile[TT][CT + 8];

    const int tid = threadIdx.x;
    const int gm  = blockIdx.x * TT;
    const int bb  = gm / SEQ;
    const int t0  = gm % SEQ;                          // TT | SEQ: no batch straddle
    const int c0  = blockIdx.y * CT;

    for (int idx = tid; idx < RR * CT; idx += 256) {
        int r = idx / CT, c = idx % CT;                // c fastest -> coalesced
        int t = t0 - HIST + r;
        float v = 0.f;
        if (t >= 0) v = u[((size_t)(bb * SEQ + t)) * DI + c0 + c];
        utile[c][r] = v;
    }
    for (int idx = tid; idx < CT * CTAP; idx += 256) {
        int c = idx / CTAP, m = idx % CTAP;
        taps_s[c][m] = taps_g[(size_t)(c0 + c) * CTAP + m];
    }
    if (tid < LTAP) wsum_s[tid] = wsum_g[tid];
    __syncthreads();

    const int lane = tid & 31;
    const int wv   = tid >> 5;
    const int cw0  = wv * 8;
    const int tg   = t0 + lane;
    const float ws = wsum_s[(tg < LTAP) ? tg : (LTAP - 1)];

    float acc[8];
    #pragma unroll
    for (int ch = 0; ch < 8; ++ch) acc[ch] = conv_b[c0 + cw0 + ch] * ws;

    for (int m = 0; m < CTAP; ++m) {
        int r = lane + HIST - m;
        #pragma unroll
        for (int ch = 0; ch < 8; ++ch)
            acc[ch] += taps_s[cw0 + ch][m] * utile[cw0 + ch][r];
    }
    #pragma unroll
    for (int ch = 0; ch < 8; ++ch) {
        float x = acc[ch];
        float s = x / (1.f + __expf(-x));              // silu
        ytile[lane][cw0 + ch] = (__bf16)s;
    }
    __syncthreads();
    {   // coalesced 16B bf16 stores
        int t = tid >> 3, c8 = tid & 7;
        v4u v = *(const v4u*)&ytile[t][c8 * 8];
        *(v4u*)(yb + ((size_t)(gm + t)) * DI + c0 + c8 * 8) = v;
    }
}

// ---------------------------------------------------------------------------
// Host-side launch.  Workspace layout (~110 MB):
//   u f32 64MB | y bf16 32MB | Win bf16 4MB | Wout bf16 4MB | w/wsum/taps
// ---------------------------------------------------------------------------
extern "C" void kernel_launch(void* const* d_in, const int* in_sizes, int n_in,
                              void* d_out, int out_size, void* d_ws, size_t ws_size,
                              hipStream_t stream) {
    const float* x      = (const float*)d_in[0];
    const float* W_in   = (const float*)d_in[1];
    const float* b_in   = (const float*)d_in[2];
    const float* conv_w = (const float*)d_in[3];
    const float* conv_b = (const float*)d_in[4];
    const float* A      = (const float*)d_in[5];
    const float* Bp     = (const float*)d_in[6];
    const float* Cp     = (const float*)d_in[7];
    const float* Dp     = (const float*)d_in[8];
    const float* W_out  = (const float*)d_in[9];
    const float* b_out  = (const float*)d_in[10];
    float* out = (float*)d_out;

    char* ws = (char*)d_ws;
    size_t off = 0;
    float*  u     = (float*)(ws + off);  off += (size_t)MTOT * DI * 4;         // 64MB
    __bf16* yb    = (__bf16*)(ws + off); off += (size_t)MTOT * DI * 2;         // 32MB
    __bf16* WinB  = (__bf16*)(ws + off); off += (size_t)DM * DI * 2;           // 4MB
    __bf16* WoutB = (__bf16*)(ws + off); off += (size_t)DI * DM * 2;           // 4MB
    float*  wv_g  = (float*)(ws + off);  off += 256;
    float*  wsum  = (float*)(ws + off);  off += 256;
    float*  taps  = (float*)(ws + off);  off += (size_t)DI * CTAP * 4;

    // 1) weights -> bf16
    int ncvt = DM * DI + DI * DM;
    cvt_w_kernel<<<(ncvt + 255) / 256, 256, 0, stream>>>(W_in, W_out, WinB, WoutB,
                                                         DM * DI, DI * DM);
    // 2) SSM impulse response + bias prefix sums
    ssm_w_kernel<<<1, 64, 0, stream>>>(A, Bp, Cp, Dp, wv_g, wsum);
    // 3) per-channel combined taps
    taps_kernel<<<(DI * CTAP + 255) / 256, 256, 0, stream>>>(conv_w, Dp, wv_g, taps);
    // 4) u = x @ W_in + b_in   (bf16 WMMA, f32 accumulate, double-buffered)
    dim3 g1(MTOT / BM, DI / BN);
    gemm_bias_kernel<true><<<g1, 256, 0, stream>>>((const void*)x, WinB, b_in, u,
                                                   MTOT, DI, DM);
    // 5) fused conv + SSM(as 67-tap causal conv) + SiLU -> bf16
    dim3 g2(MTOT / 32, DI / 64);
    convscan_kernel<<<g2, 256, 0, stream>>>(u, conv_b, taps, wsum, yb);
    // 6) out = y @ W_out + b_out
    dim3 g3(MTOT / BM, DM / BN);
    gemm_bias_kernel<false><<<g3, 256, 0, stream>>>((const void*)yb, WoutB, b_out, out,
                                                    MTOT, DM, DI);
}
